// Recurrence_4063039062398
// MI455X (gfx1250) — matrix operand order repaired and checked by reading, not compiled, not run on hardware
//
#include <hip/hip_runtime.h>
#include <math.h>

typedef __attribute__((ext_vector_type(16))) _Float16 v16h;
typedef __attribute__((ext_vector_type(8)))  _Float16 h8;
typedef __attribute__((ext_vector_type(8)))  float    v8f;

#define TSTEPS  32
#define NB      256
#define HID     64
#define OBS_LEN 128
#define NSUB    6
#define SUBDIM  3
#define OBS_V   16
#define IN_COLS 147          // 128 obs + 18 sub + 1 act
#define G3      192          // 3*HID gates
#define KQ      2048         // one-hot K: 128 slots * 16 values
#define GRU_IN  8384
#define KOBS    8192
#define STATE   78
#define MKLEN   (NSUB*G3)    // per-n M block: 6*192

static __device__ __forceinline__ float sigf(float x){ return 1.0f/(1.0f+expf(-x)); }

static __device__ __forceinline__ v8f wmma_f16(v16h a, v16h b, v8f c){
  return __builtin_amdgcn_wmma_f32_16x16x32_f16(false, a, false, b, (short)0, c, false, false);
}

// ---- small prep kernels -------------------------------------------------
__global__ void k_cvt(const float* __restrict__ src, _Float16* __restrict__ dst, int n){
  int i = blockIdx.x*blockDim.x + threadIdx.x;
  if (i < n) dst[i] = (_Float16)src[i];
}

// w_ih[:, 8192:] slice -> f16 [192][192]
__global__ void k_cvt_wihr(const float* __restrict__ wih, _Float16* __restrict__ dst){
  int i = blockIdx.x*blockDim.x + threadIdx.x;
  if (i < G3*G3){
    int g = i / G3, k = i % G3;
    dst[i] = (_Float16)wih[(size_t)g*GRU_IN + KOBS + k];
  }
}

// M[n][s][k] = subtask_emb[ sub_idx0[n, s*3 + (k>>6)] ][ k&63 ]
__global__ void k_build_M(const float* __restrict__ inputs, const float* __restrict__ semb,
                          float* __restrict__ Mbuf){
  int i = blockIdx.x*blockDim.x + threadIdx.x;
  if (i < NB*MKLEN){
    int n = i / MKLEN, r = i % MKLEN;
    int s = r / G3, k = r % G3;
    int d = k >> 6, h = k & 63;
    int idx = (int)inputs[n*IN_COLS + OBS_LEN + s*SUBDIM + d];
    Mbuf[i] = semb[idx*HID + h];
  }
}

__global__ void k_init_state(const float* __restrict__ hx, float* __restrict__ hbuf,
                             _Float16* __restrict__ h16, float* __restrict__ pbuf){
  int n = threadIdx.x;
  bool nz = false;
  for (int j = 0; j < STATE; ++j) nz |= (hx[n*STATE + j] != 0.0f);
  for (int j = 0; j < HID; ++j){
    float v = hx[n*STATE + 2 + j];
    hbuf[n*HID + j] = v; h16[n*HID + j] = (_Float16)v;
  }
  for (int s = 0; s < NSUB; ++s) pbuf[n*NSUB + s] = hx[n*STATE + 2 + HID + 6 + s];
  if (!nz) pbuf[n*NSUB + 0] = 1.0f;   // new episode: p0[:,0] = 1
}

// CE[o, ch, v] = sum_h conv_w[o, ch*64+h] * obs_emb[v, h]   (128*2*16)
__global__ void k_ce(const float* __restrict__ convw, const float* __restrict__ oemb,
                     float* __restrict__ CE){
  int i = blockIdx.x*blockDim.x + threadIdx.x;
  if (i < 4096){
    int o = i >> 5, ch = (i >> 4) & 1, v = i & 15;
    const float* cw = convw + o*(2*HID) + ch*HID;
    const float* ev = oemb + v*HID;
    float s = 0.f;
    for (int h = 0; h < HID; ++h) s += cw[h]*ev[h];
    CE[i] = s;
  }
}

// Q[g, slot*16+v] = sum_o w_ih[g, o*64 + (slot&63)] * CE[o, slot>>6, v]
__global__ void k_q(const float* __restrict__ wih, const float* __restrict__ CE,
                    _Float16* __restrict__ Qtab){
  int i = blockIdx.x*blockDim.x + threadIdx.x;
  if (i < G3*KQ){
    int g = i >> 11, kq = i & (KQ-1);
    int slot = kq >> 4, v = kq & 15;
    int ch = slot >> 6, p = slot & 63;
    const float* wr = wih + (size_t)g*GRU_IN + p;   // stride 64 over o
    const float* ce = CE + ch*16 + v;               // stride 32 over o
    float s = 0.f;
    for (int o = 0; o < 2*HID; ++o) s += wr[o*HID]*ce[o*32];
    Qtab[i] = (_Float16)s;
  }
}

// b_eff[g] = b_ih[g] + sum_o conv_b[o] * sum_p w_ih[g, o*64+p]
__global__ void k_beff(const float* __restrict__ wih, const float* __restrict__ convb,
                       const float* __restrict__ bih, float* __restrict__ beff){
  int g = threadIdx.x;
  if (g < G3){
    float b = bih[g];
    for (int o = 0; o < 2*HID; ++o){
      const float* wr = wih + (size_t)g*GRU_IN + o*HID;
      float rs = 0.f;
      for (int pp = 0; pp < 64; ++pp) rs += wr[pp];
      b += convb[o]*rs;
    }
    beff[g] = b;
  }
}

// ---- big precomputable GEMM: gxobs[T*N][192] = OneHot(obs) @ Q^T --------
// One block (4 waves) = one 16-row m-tile covering all 192 output cols.
// Each wave: 3 g-tiles; A fragment is built from 2 index bytes + LDS LUT.
__global__ void __launch_bounds__(128) k_gx_gemm(
    const float* __restrict__ inputs, const _Float16* __restrict__ Qtab,
    const float* __restrict__ beff, float* __restrict__ gxobs){
  __shared__ alignas(16) unsigned char idx8[16*128];
  __shared__ alignas(16) _Float16 lut[24*8];     // one-hot 8-wide patterns
  const int tid  = threadIdx.x;
  const int lane = tid & 31, wid = tid >> 5;
  const int m0 = blockIdx.x * 16;
  for (int j = tid; j < 2048; j += 128)
    idx8[j] = (unsigned char)(int)inputs[(m0 + (j >> 7))*IN_COLS + (j & 127)];
  for (int j = tid; j < 192; j += 128)
    lut[j] = ((j >> 3) - 8 == (j & 7)) ? (_Float16)1.0f : (_Float16)0.0f;
  __syncthreads();

  const int half = lane >> 4;
  const unsigned char* idxrow = &idx8[(lane & 15)*128];
  const _Float16* lutb = lut + (8 - half*8)*8;   // entry u = s + 8 - half*8
  const _Float16* q0 = Qtab + (size_t)((wid*3 + 0)*16 + (lane & 15))*KQ + half*8;
  const _Float16* q1 = q0 + (size_t)16*KQ;
  const _Float16* q2 = q1 + (size_t)16*KQ;
  v8f acc0 = {}, acc1 = {}, acc2 = {};
  #pragma unroll 4
  for (int kb = 0; kb < KQ/32; ++kb){
    unsigned sv = *reinterpret_cast<const unsigned short*>(&idxrow[kb*2]);
    v16h a, b0, b1, b2;
    reinterpret_cast<h8*>(&a)[0] = *reinterpret_cast<const h8*>(lutb + (sv & 255u)*8);
    reinterpret_cast<h8*>(&a)[1] = *reinterpret_cast<const h8*>(lutb + (sv >> 8)*8);
    const int ko = kb*32;
    reinterpret_cast<h8*>(&b0)[0] = *reinterpret_cast<const h8*>(q0 + ko);
    reinterpret_cast<h8*>(&b0)[1] = *reinterpret_cast<const h8*>(q0 + ko + 16);
    reinterpret_cast<h8*>(&b1)[0] = *reinterpret_cast<const h8*>(q1 + ko);
    reinterpret_cast<h8*>(&b1)[1] = *reinterpret_cast<const h8*>(q1 + ko + 16);
    reinterpret_cast<h8*>(&b2)[0] = *reinterpret_cast<const h8*>(q2 + ko);
    reinterpret_cast<h8*>(&b2)[1] = *reinterpret_cast<const h8*>(q2 + ko + 16);
    acc0 = wmma_f16(a, b0, acc0);
    acc1 = wmma_f16(a, b1, acc1);
    acc2 = wmma_f16(a, b2, acc2);
  }
  #pragma unroll
  for (int gt = 0; gt < 3; ++gt){
    v8f acc = (gt == 0) ? acc0 : ((gt == 1) ? acc1 : acc2);
    const int colg = (wid*3 + gt)*16 + (lane & 15);
    const float be = beff[colg];
    #pragma unroll
    for (int r = 0; r < 8; ++r){
      int row = m0 + r + half*8;
      gxobs[(size_t)row*G3 + colg] = acc[r] + be;
    }
  }
}

// ---- persistent sequential scan (1 block, 16 waves, WMMA recurrent GEMMs)
__global__ void __launch_bounds__(512) k_scan(
    const float* __restrict__ inputs, const float* __restrict__ gxobs,
    const float* __restrict__ Mbuf,
    const _Float16* __restrict__ wihr, const _Float16* __restrict__ whh,
    const float* __restrict__ bhh,
    const float* __restrict__ phw, const float* __restrict__ phb,
    const float* __restrict__ crw, const float* __restrict__ crb,
    float* __restrict__ hbuf, _Float16* __restrict__ h16,
    _Float16* __restrict__ rv16,
    float* __restrict__ gxb, float* __restrict__ ghb,
    float* __restrict__ pbuf, float* __restrict__ out)
{
  const int tid = threadIdx.x;
  const int lane = tid & 31, wid = tid >> 5;
  for (int t = 0; t < TSTEPS; ++t){
    // Phase 1: r_vec = p @ M, f16
    for (int i = tid; i < NB*G3; i += 512){
      int n = i / G3, k = i % G3;
      const float* pm = Mbuf + n*MKLEN + k;
      const float* pp = pbuf + n*NSUB;
      float s = 0.f;
      #pragma unroll
      for (int ss = 0; ss < NSUB; ++ss) s += pp[ss]*pm[ss*G3];
      rv16[i] = (_Float16)s;
    }
    __threadfence(); __syncthreads();
    // Phase 2: gx = gxobs + rv@wihr.T ; gh = h@whh.T + b_hh   (16x16 WMMA tiles)
    for (int tile = wid; tile < 192; tile += 16){
      const int m0 = (tile / 12)*16, g0 = (tile % 12)*16;
      const int half = lane >> 4;
      const int arow = m0 + (lane & 15);
      const int bcol = g0 + (lane & 15);
      v8f acc = {};
      #pragma unroll
      for (int kb = 0; kb < 6; ++kb){
        const _Float16* ap = rv16 + arow*G3 + kb*32 + half*8;
        v16h a, b;
        reinterpret_cast<h8*>(&a)[0] = *reinterpret_cast<const h8*>(ap);
        reinterpret_cast<h8*>(&a)[1] = *reinterpret_cast<const h8*>(ap + 16);
        const _Float16* bp = wihr + bcol*G3 + kb*32 + half*8;
        reinterpret_cast<h8*>(&b)[0] = *reinterpret_cast<const h8*>(bp);
        reinterpret_cast<h8*>(&b)[1] = *reinterpret_cast<const h8*>(bp + 16);
        acc = wmma_f16(a, b, acc);
      }
      #pragma unroll
      for (int r = 0; r < 8; ++r){
        int row = m0 + r + half*8;
        gxb[row*G3 + bcol] = acc[r] + gxobs[((size_t)t*NB + row)*G3 + bcol];
      }
      v8f acc2 = {};
      #pragma unroll
      for (int kb = 0; kb < 2; ++kb){
        const _Float16* ap = h16 + arow*HID + kb*32 + half*8;
        v16h a, b;
        reinterpret_cast<h8*>(&a)[0] = *reinterpret_cast<const h8*>(ap);
        reinterpret_cast<h8*>(&a)[1] = *reinterpret_cast<const h8*>(ap + 16);
        const _Float16* bp = whh + bcol*HID + kb*32 + half*8;
        reinterpret_cast<h8*>(&b)[0] = *reinterpret_cast<const h8*>(bp);
        reinterpret_cast<h8*>(&b)[1] = *reinterpret_cast<const h8*>(bp + 16);
        acc2 = wmma_f16(a, b, acc2);
      }
      const float bh = bhh[bcol];
      #pragma unroll
      for (int r = 0; r < 8; ++r){
        int row = m0 + r + half*8;
        ghb[row*G3 + bcol] = acc2[r] + bh;
      }
    }
    __threadfence(); __syncthreads();
    // Phase 3: GRU gates -> h_new
    for (int i = tid; i < NB*HID; i += 512){
      int n = i >> 6, j = i & 63;
      float r  = sigf(gxb[n*G3 + j]      + ghb[n*G3 + j]);
      float z  = sigf(gxb[n*G3 + 64 + j] + ghb[n*G3 + 64 + j]);
      float nn = tanhf(gxb[n*G3 + 128 + j] + r*ghb[n*G3 + 128 + j]);
      float hn = (1.f - z)*nn + z*hbuf[i];
      hbuf[i] = hn;
      h16[i]  = (_Float16)hn;
      out[((size_t)t*NB + n)*STATE + 2 + j] = hn;
    }
    __threadfence(); __syncthreads();
    // Phase 4: per-row scalar tail (critic, phi, pointer update, outputs)
    if (tid < NB){
      int n = tid;
      float ci = phb[0], v = crb[0];
      for (int j = 0; j < HID; ++j){
        float hv = hbuf[n*HID + j];
        ci += hv*phw[j];
        v  += hv*crw[j];
      }
      float c = sigf(ci);
      float pv[NSUB];
      #pragma unroll
      for (int s = 0; s < NSUB; ++s) pv[s] = pbuf[n*NSUB + s];
      float last = pv[NSUB-1];
      float pn[NSUB], sum = 0.f;
      #pragma unroll
      for (int s = 0; s < NSUB; ++s){
        float p1 = (s == 0) ? 0.f : pv[s-1];
        float p2 = p1 + pv[s]*last;
        pn[s] = (1.f - c)*pv[s] + c*p2;
        sum += pn[s];
      }
      float inv = 1.f / sum;
      size_t base = ((size_t)t*NB + n)*STATE;
      out[base + 0] = inputs[((size_t)t*NB + n)*IN_COLS + (IN_COLS-1)];
      out[base + 1] = v;
      #pragma unroll
      for (int s = 0; s < NSUB; ++s){
        out[base + 2 + HID + s]        = pn[s]*inv;  // probs
        out[base + 2 + HID + NSUB + s] = pn[s];      // p_new
        pbuf[n*NSUB + s] = pn[s];
      }
    }
    __threadfence(); __syncthreads();
  }
  // second output: hx_out[-1:]
  for (int i = tid; i < NB*STATE; i += 512){
    out[(size_t)TSTEPS*NB*STATE + i] = out[((size_t)(TSTEPS-1)*NB)*STATE + i];
  }
}

extern "C" void kernel_launch(void* const* d_in, const int* in_sizes, int n_in,
                              void* d_out, int out_size, void* d_ws, size_t ws_size,
                              hipStream_t stream) {
  (void)in_sizes; (void)n_in; (void)out_size; (void)ws_size;
  const float* inputs = (const float*)d_in[0];
  const float* hx     = (const float*)d_in[1];
  const float* semb   = (const float*)d_in[2];
  const float* oemb   = (const float*)d_in[3];
  const float* convw  = (const float*)d_in[4];
  const float* convb  = (const float*)d_in[5];
  const float* wih    = (const float*)d_in[6];
  const float* whh    = (const float*)d_in[7];
  const float* bih    = (const float*)d_in[8];
  const float* bhh    = (const float*)d_in[9];
  const float* crw    = (const float*)d_in[10];
  const float* crb    = (const float*)d_in[11];
  const float* phw    = (const float*)d_in[12];
  const float* phb    = (const float*)d_in[13];
  float* out = (float*)d_out;

  char* ws = (char*)d_ws;
  size_t off = 0;
  auto take = [&](size_t bytes)->char*{
    char* p = ws + off; off = (off + bytes + 255) & ~(size_t)255; return p;
  };
  _Float16* whh16  = (_Float16*)take((size_t)G3*HID*2);
  _Float16* wihr16 = (_Float16*)take((size_t)G3*G3*2);
  float*    CE     = (float*)   take((size_t)4096*4);
  _Float16* Qtab   = (_Float16*)take((size_t)G3*KQ*2);
  float*    beff   = (float*)   take((size_t)G3*4);
  float*    Mbuf   = (float*)   take((size_t)NB*MKLEN*4);
  float*    gxobs  = (float*)   take((size_t)TSTEPS*NB*G3*4);
  float*    hbuf   = (float*)   take((size_t)NB*HID*4);
  _Float16* h16    = (_Float16*)take((size_t)NB*HID*2);
  _Float16* rv16   = (_Float16*)take((size_t)NB*G3*2);
  float*    gxb    = (float*)   take((size_t)NB*G3*4);
  float*    ghb    = (float*)   take((size_t)NB*G3*4);
  float*    pbuf   = (float*)   take((size_t)NB*NSUB*4);

  k_cvt<<<(G3*HID + 255)/256, 256, 0, stream>>>(whh, whh16, G3*HID);
  k_cvt_wihr<<<(G3*G3 + 255)/256, 256, 0, stream>>>(wih, wihr16);
  k_build_M<<<(NB*MKLEN + 255)/256, 256, 0, stream>>>(inputs, semb, Mbuf);
  k_init_state<<<1, 256, 0, stream>>>(hx, hbuf, h16, pbuf);
  k_ce<<<16, 256, 0, stream>>>(convw, oemb, CE);
  k_q<<<(G3*KQ)/256, 256, 0, stream>>>(wih, CE, Qtab);
  k_beff<<<1, 256, 0, stream>>>(wih, convb, bih, beff);
  k_gx_gemm<<<(TSTEPS*NB)/16, 128, 0, stream>>>(inputs, Qtab, beff, gxobs);
  k_scan<<<1, 512, 0, stream>>>(inputs, gxobs, Mbuf, wihr16, whh16, bhh,
                                phw, phb, crw, crb,
                                hbuf, h16, rv16, gxb, ghb, pbuf, out);
}
